// LDA_17523466568092
// MI455X (gfx1250) — matrix operand order repaired and checked by reading, not compiled, not run on hardware
//
#include <hip/hip_runtime.h>
#include <hip/hip_bf16.h>

// ---------------------------------------------------------------------------
// MI455X (gfx1250) implementation.
//  - complex 3x3 convs  -> implicit GEMM on v_wmma_f32_16x16x32_bf16 (wave32),
//                          weights staged to LDS via async global->LDS loads,
//                          K ordered as ((p*3+q)*2+ri)*Cin+c so the unrolled
//                          inner loop has no per-element index decode.
//  - FFT2/IFFT2         -> DFT matrices (fp32 VALU GEMM; ~1% of total FLOPs)
//  - everything else    -> elementwise VALU kernels
// ---------------------------------------------------------------------------

#define BB 8
#define HH 160
#define WW 180
#define HW (HH * WW)          // 28800
#define NP (BB * HW)          // 230400 (divisible by 16)
#define NCH 32
#define CAP (BB * NCH * HW)   // 7372800
#define DELTA_ 0.01f

#define AS1 __attribute__((address_space(1)))
#define AS3 __attribute__((address_space(3)))

#if __has_builtin(__builtin_amdgcn_global_load_async_to_lds_b128)
#define ASYNC_LDS 1
#else
#define ASYNC_LDS 0
#endif

typedef __attribute__((ext_vector_type(16))) __bf16 v16bf;
typedef __attribute__((ext_vector_type(8)))  float  v8f;
typedef int v4i __attribute__((vector_size(16)));   // matches builtin param type

__device__ __forceinline__ float act_f(float x) {
    return (fabsf(x) > DELTA_) ? (x > 0.f ? x : 0.f)
                               : (x * x * (1.f / (4.f * DELTA_)) + 0.5f * x + DELTA_ * 0.25f);
}
__device__ __forceinline__ float act_der_f(float x) {
    return (fabsf(x) > DELTA_) ? (x > 0.f ? 1.f : 0.f)
                               : (x * (1.f / (2.f * DELTA_)) + 0.5f);
}

// ---------------------------------------------------------------------------
// Pack complex conv weights into the WMMA A-matrix VGPR layout (bf16).
// Matrix rows m: [0,Cout) -> real outputs, [Cout,2*Cout) -> imag outputs.
// K index = ((p*3 + q)*2 + ri)*Cin + c   (ri: 0=real input, 1=imag input).
// A order used here (16x32 bf16, contiguous K, not the ISA table order): lane L holds M = L%16;
//   element e -> K = 16*(L/16) + e   within the 32-K step.
// apack layout: [Mtiles][Ksteps][32 lanes][16 elems]
// ---------------------------------------------------------------------------
__global__ void pack_weights_kernel(const float* __restrict__ kr,
                                    const float* __restrict__ ki,
                                    __bf16* __restrict__ apack,
                                    int Cin, int Cout, int Ksteps, int Mtiles,
                                    int transposed)
{
    int idx = blockIdx.x * blockDim.x + threadIdx.x;
    int total = Mtiles * Ksteps * 32 * 16;
    if (idx >= total) return;
    int e    = idx & 15;
    int lane = (idx >> 4) & 31;
    int rest = idx >> 9;
    int ks   = rest % Ksteps;
    int mt   = rest / Ksteps;
    int half = lane >> 4;
    int m    = mt * 16 + (lane & 15);
    int kk   = ks * 32 + half * 16 + e;
    int Ktot = 18 * Cin;
    float v = 0.f;
    if (m < 2 * Cout && kk < Ktot) {
        int c   = kk % Cin;
        int t   = kk / Cin;
        int ri  = t & 1;
        int pq  = t >> 1;
        int p   = pq / 3;
        int q   = pq - p * 3;
        int oc  = (m < Cout) ? m : m - Cout;
        int widx;
        if (!transposed) widx = ((oc * Cin + c) * 3 + p) * 3 + q;
        else             widx = ((c * Cout + oc) * 3 + (2 - p)) * 3 + (2 - q);
        float wr = kr[widx], wi = ki[widx];
        v = (m < Cout) ? (ri == 0 ? wr : -wi)
                       : (ri == 0 ? wi :  wr);
    }
    apack[idx] = (__bf16)v;
}

// ---------------------------------------------------------------------------
// Implicit-GEMM complex conv via WMMA bf16.
// One wave = one 16-column tile of N = B*H*W, MTILES accumulators.
// B fragment (32x16 bf16): lane L holds column N = L%16, element e ->
//   K = 16*(L/16) + e within the 32-K step.
// With the (p,q,ri)-major K order and CIN==32, one K-step is one (p,q,ri)
// combo and e indexes the input channel directly -> no index decode.
// Packed weights are staged to LDS with async global->LDS (ASYNCcnt) and
// read back as A fragments with ds_load.
// ---------------------------------------------------------------------------
template<int CIN, int KSTEPS, int MTILES>
__global__ void __launch_bounds__(256)
conv_wmma_kernel(const float* __restrict__ in_r, const float* __restrict__ in_i,
                 const __bf16* __restrict__ apack,
                 float* __restrict__ out_r, float* __restrict__ out_i,
                 int Cout, int actMode)
{
    extern __shared__ __align__(32) char lds_raw[];
    __bf16* aplds = (__bf16*)lds_raw;

    constexpr int APELEMS = MTILES * KSTEPS * 512;
    constexpr int NVEC    = APELEMS / 8;          // 16-byte vectors

    // ---- stage packed weights into LDS (shared by all 8 waves) ----
    for (int i = threadIdx.x; i < NVEC; i += 256) {
#if ASYNC_LDS
        __builtin_amdgcn_global_load_async_to_lds_b128(
            (AS1 v4i*)(apack + (size_t)i * 8),
            (AS3 v4i*)(aplds + (size_t)i * 8), 0, 0);
#else
        ((int4*)aplds)[i] = ((const int4*)apack)[i];
#endif
    }
#if ASYNC_LDS
#if __has_builtin(__builtin_amdgcn_s_wait_asynccnt)
    __builtin_amdgcn_s_wait_asynccnt(0);
#else
    asm volatile("s_wait_asynccnt 0" ::: "memory");
#endif
#endif
    __syncthreads();

    const int lane = threadIdx.x & 31;
    const int hf   = lane >> 4;
    const int wv   = threadIdx.x >> 5;
    const int n0   = (blockIdx.x * (blockDim.x >> 5) + wv) * 16;
    if (n0 >= NP) return;                     // wave-uniform; grid is exact

    const int ncol = n0 + (lane & 15);
    const int b  = ncol / HW;
    const int hw = ncol - b * HW;
    const int h  = hw / WW;
    const int w  = hw - h * WW;

    __builtin_prefetch(in_r + (size_t)b * CIN * HW + (size_t)h * WW + w, 0, 1);
    __builtin_prefetch(in_i + (size_t)b * CIN * HW + (size_t)h * WW + w, 0, 1);

    v8f zero = {0.f, 0.f, 0.f, 0.f, 0.f, 0.f, 0.f, 0.f};
    v8f acc[MTILES];
    #pragma unroll
    for (int mt = 0; mt < MTILES; ++mt) acc[mt] = zero;

    #pragma unroll
    for (int ks = 0; ks < KSTEPS; ++ks) {
        v16bf bfrag;
        if constexpr (CIN == 32) {
            // one (p,q,ri) per K-step; channel c = hf*16 + e
            const int p  = ks / 6;
            const int q  = (ks / 2) % 3;
            const int ri = ks & 1;
            const int hh = h + p - 1;
            const int w2 = w + q - 1;
            const bool ok = (hh >= 0) && (hh < HH) && (w2 >= 0) && (w2 < WW);
            const float* src = ri ? in_i : in_r;
            const size_t base = (size_t)b * CIN * HW + (size_t)(hf * 16) * HW
                              + (size_t)hh * WW + w2;
            #pragma unroll
            for (int e = 0; e < 16; ++e) {
                float v = ok ? src[base + (size_t)e * HW] : 0.f;
                bfrag[e] = (__bf16)v;
            }
        } else {
            // CIN == 1: Ktot = 18, single K-step, tiny decode
            #pragma unroll
            for (int e = 0; e < 16; ++e) {
                const int kk = hf * 16 + e;
                float v = 0.f;
                if (kk < 18) {
                    const int ri = kk & 1;
                    const int pq = kk >> 1;
                    const int p  = pq / 3;
                    const int q  = pq - 3 * p;
                    const int hh = h + p - 1;
                    const int w2 = w + q - 1;
                    if (hh >= 0 && hh < HH && w2 >= 0 && w2 < WW)
                        v = (ri ? in_i : in_r)[(size_t)b * HW + (size_t)hh * WW + w2];
                }
                bfrag[e] = (__bf16)v;
            }
        }

        #pragma unroll
        for (int mt = 0; mt < MTILES; ++mt) {
            v16bf afrag = *(const v16bf*)(aplds +
                (((size_t)mt * KSTEPS + ks) * 32 + lane) * 16);
            acc[mt] = __builtin_amdgcn_wmma_f32_16x16x32_bf16(
                false, afrag, false, bfrag, (short)0, acc[mt], false, false);
        }
    }

    // D layout: lane L -> N = L%16 ; VGPR v -> M = v + 8*(L/16)
    #pragma unroll
    for (int mt = 0; mt < MTILES; ++mt) {
        #pragma unroll
        for (int v = 0; v < 8; ++v) {
            const int m = mt * 16 + hf * 8 + v;
            if (m < 2 * Cout) {
                float x = acc[mt][v];
                if (actMode) x = act_f(x);
                const int oc = (m < Cout) ? m : m - Cout;
                float* dst = (m < Cout) ? out_r : out_i;
                dst[((size_t)(b * Cout + oc) * HH + h) * WW + w] = x;
            }
        }
    }
}

// ---------------------------------------------------------------------------
// DFT matrix builder: F[u,n] = exp(-2*pi*i*u*n/N); IF = conj(F)/N.
// ---------------------------------------------------------------------------
__global__ void dft_kernel(float* Fr, float* Fi, float* IFr, float* IFi, int N)
{
    int i = blockIdx.x * blockDim.x + threadIdx.x;
    if (i >= N * N) return;
    int u = i / N;
    int n = i - u * N;
    int m = (int)(((long long)u * (long long)n) % (long long)N);
    float a = -6.283185307179586f * (float)m / (float)N;
    float c = cosf(a), s = sinf(a);
    Fr[i] = c;             Fi[i] = s;
    IFr[i] = c / (float)N; IFi[i] = -s / (float)N;
}

// Batched complex GEMM C = A x B (fp32 VALU; DFT path is ~1% of FLOPs).
__global__ void cgemm_kernel(const float* __restrict__ Ar, const float* __restrict__ Ai,
                             const float* __restrict__ Br, const float* __restrict__ Bi,
                             float* __restrict__ Cr, float* __restrict__ Ci,
                             int M, int K, int N, int as_, int bs_, int cs_)
{
    const int bz = blockIdx.z;
    Ar += (size_t)bz * as_; Ai += (size_t)bz * as_;
    Br += (size_t)bz * bs_; Bi += (size_t)bz * bs_;
    Cr += (size_t)bz * cs_; Ci += (size_t)bz * cs_;
    const int col = blockIdx.x * 16 + threadIdx.x;
    const int row = blockIdx.y * 16 + threadIdx.y;
    if (row >= M || col >= N) return;
    float sr = 0.f, si = 0.f;
    for (int k = 0; k < K; ++k) {
        const float ar = Ar[(size_t)row * K + k];
        const float ai = Ai[(size_t)row * K + k];
        const float br = Br[(size_t)k * N + col];
        const float bi = Bi[(size_t)k * N + col];
        sr = fmaf(ar, br, sr); sr = fmaf(-ai, bi, sr);
        si = fmaf(ar, bi, si); si = fmaf(ai, br, si);
    }
    Cr[(size_t)row * N + col] = sr;
    Ci[(size_t)row * N + col] = si;
}

// Z = (Y*mask - k)*mask  (in place on Y)
__global__ void maskres_kernel(float* __restrict__ yr, float* __restrict__ yi,
                               const float* __restrict__ mask,
                               const float* __restrict__ kre,
                               const float* __restrict__ kim, int n)
{
    int i = blockIdx.x * blockDim.x + threadIdx.x;
    if (i >= n) return;
    int hw = i % HW;
    float m = mask[hw];
    float zr = yr[i] * m - kre[i];
    float zi = yi[i] * m - kim[i];
    yr[i] = zr * m;
    yi[i] = zi * m;
}

// channel-norm shrink: g = w4 / max(norm, thr-ish) per the reference formula
__global__ void norm_scale_kernel(const float* __restrict__ w4r, const float* __restrict__ w4i,
                                  float* __restrict__ g0r, float* __restrict__ g0i,
                                  const float* __restrict__ thrp, int np)
{
    int i = blockIdx.x * blockDim.x + threadIdx.x;
    if (i >= np) return;
    int b  = i / HW;
    int hw = i - b * HW;
    size_t base = (size_t)b * NCH * HW + hw;
    float s = 0.f;
    for (int c = 0; c < NCH; ++c) {
        float wr = w4r[base + (size_t)c * HW];
        float wi = w4i[base + (size_t)c * HW];
        s = fmaf(wr, wr, s);
        s = fmaf(wi, wi, s);
    }
    float norm = sqrtf(s);
    float thr  = thrp[0] * 1.0f;   // GAMMA = 1
    float scale = (norm > thr) ? (1.f / fmaxf(norm, 1e-12f)) : (1.f / thr);
    for (int c = 0; c < NCH; ++c) {
        g0r[base + (size_t)c * HW] = w4r[base + (size_t)c * HW] * scale;
        g0i[base + (size_t)c * HW] = w4i[base + (size_t)c * HW] * scale;
    }
}

// g <- g * complex(act_der(a_r), act_der(a_i))   (complex multiply, in place)
__global__ void actder_mul_kernel(float* __restrict__ gr, float* __restrict__ gi,
                                  const float* __restrict__ ar, const float* __restrict__ ai,
                                  int n)
{
    int i = blockIdx.x * blockDim.x + threadIdx.x;
    if (i >= n) return;
    float dr = act_der_f(ar[i]);
    float di = act_der_f(ai[i]);
    float r  = gr[i], im = gi[i];
    gr[i] = r * dr - im * di;
    gi[i] = r * di + im * dr;
}

__global__ void update_kernel(float* __restrict__ xr, float* __restrict__ xi,
                              const float* __restrict__ gfr, const float* __restrict__ gfi,
                              const float* __restrict__ grr, const float* __restrict__ gri,
                              const float* __restrict__ alphas, const float* __restrict__ betas,
                              int phase, int n)
{
    int i = blockIdx.x * blockDim.x + threadIdx.x;
    if (i >= n) return;
    float a = alphas[phase], b = betas[phase];
    xr[i] -= a * gfr[i] + b * grr[i];
    xi[i] -= a * gfi[i] + b * gri[i];
}

__global__ void split_ri_kernel(const float* __restrict__ src,
                                float* __restrict__ r, float* __restrict__ im, int n)
{
    int i = blockIdx.x * blockDim.x + threadIdx.x;
    if (i >= n) return;
    int b  = i / HW;
    int hw = i - b * HW;
    r[i]  = src[(size_t)(b * 2 + 0) * HW + hw];
    im[i] = src[(size_t)(b * 2 + 1) * HW + hw];
}

__global__ void pack_out_kernel(const float* __restrict__ x1r, const float* __restrict__ x1i,
                                const float* __restrict__ x2r, const float* __restrict__ x2i,
                                float* __restrict__ out, int n)
{
    int i = blockIdx.x * blockDim.x + threadIdx.x;
    if (i >= n) return;
    int b  = i / HW;
    int hw = i - b * HW;
    out[(size_t)(b * 4 + 0) * HW + hw] = x1r[i];
    out[(size_t)(b * 4 + 1) * HW + hw] = x1i[i];
    out[(size_t)(b * 4 + 2) * HW + hw] = x2r[i];
    out[(size_t)(b * 4 + 3) * HW + hw] = x2i[i];
}

// ---------------------------------------------------------------------------

extern "C" void kernel_launch(void* const* d_in, const int* in_sizes, int n_in,
                              void* d_out, int out_size, void* d_ws, size_t ws_size,
                              hipStream_t stream)
{
    (void)in_sizes; (void)n_in; (void)out_size; (void)ws_size;

    const float* x1in = (const float*)d_in[0];
    const float* x2in = (const float*)d_in[1];
    const float* k1in = (const float*)d_in[2];
    const float* k2in = (const float*)d_in[3];
    const float* mask = (const float*)d_in[4];
    const float* convs[16];
    for (int i = 0; i < 16; ++i) convs[i] = (const float*)d_in[5 + i];
    const float* thr1 = (const float*)d_in[21];
    const float* thr2 = (const float*)d_in[22];
    const float* al1  = (const float*)d_in[23];
    const float* al2  = (const float*)d_in[24];
    const float* be1  = (const float*)d_in[25];
    const float* be2  = (const float*)d_in[26];

    // ---- workspace carve (256B-aligned bump allocator) ----
    char* p = (char*)d_ws;
    auto alloc = [&](size_t bytes) -> void* {
        void* r = (void*)p;
        p += (bytes + 255) & ~(size_t)255;
        return r;
    };
    auto allocF = [&](size_t n) -> float* { return (float*)alloc(n * sizeof(float)); };

    float *x1r = allocF(NP), *x1i = allocF(NP), *x2r = allocF(NP), *x2i = allocF(NP);
    float *k1r = allocF(NP), *k1i = allocF(NP), *k2r = allocF(NP), *k2i = allocF(NP);
    float *FHr = allocF(HH * HH), *FHi = allocF(HH * HH);
    float *IFHr = allocF(HH * HH), *IFHi = allocF(HH * HH);
    float *FWr = allocF(WW * WW), *FWi = allocF(WW * WW);
    float *IFWr = allocF(WW * WW), *IFWi = allocF(WW * WW);
    float *t1r = allocF(NP), *t1i = allocF(NP), *t2r = allocF(NP), *t2i = allocF(NP);
    float *a1r = allocF(CAP), *a1i = allocF(CAP);
    float *a2r = allocF(CAP), *a2i = allocF(CAP);
    float *a3r = allocF(CAP), *a3i = allocF(CAP);
    float *w4r = allocF(CAP), *w4i = allocF(CAP);
    float *g0r = allocF(CAP), *g0i = allocF(CAP);
    float *g1r = allocF(CAP), *g1i = allocF(CAP);
    float *grr = allocF(NP), *gri = allocF(NP);
    const size_t APSLOT = (size_t)4 * 18 * 32 * 16;   // max packed-matrix elems
    __bf16* apack[16];
    for (int i = 0; i < 16; ++i) apack[i] = (__bf16*)alloc(APSLOT * sizeof(__bf16));

    // ---- one-time (per call) setup ----
    split_ri_kernel<<<NP / 256, 256, 0, stream>>>(x1in, x1r, x1i, NP);
    split_ri_kernel<<<NP / 256, 256, 0, stream>>>(x2in, x2r, x2i, NP);
    split_ri_kernel<<<NP / 256, 256, 0, stream>>>(k1in, k1r, k1i, NP);
    split_ri_kernel<<<NP / 256, 256, 0, stream>>>(k2in, k2r, k2i, NP);
    dft_kernel<<<(HH * HH + 255) / 256, 256, 0, stream>>>(FHr, FHi, IFHr, IFHi, HH);
    dft_kernel<<<(WW * WW + 255) / 256, 256, 0, stream>>>(FWr, FWi, IFWr, IFWi, WW);

    auto launch_pack = [&](const float* kr, const float* ki, __bf16* dst,
                           int cin, int cout, int ks, int mt, int tr) {
        int total = mt * ks * 32 * 16;
        pack_weights_kernel<<<(total + 255) / 256, 256, 0, stream>>>(
            kr, ki, dst, cin, cout, ks, mt, tr);
    };
    for (int s = 0; s < 2; ++s) {
        int wb = 8 * s;
        launch_pack(convs[wb + 0], convs[wb + 1], apack[8 * s + 0],  1, 32,  1, 4, 0); // fwd1
        launch_pack(convs[wb + 2], convs[wb + 3], apack[8 * s + 1], 32, 32, 18, 4, 0); // fwd2
        launch_pack(convs[wb + 4], convs[wb + 5], apack[8 * s + 2], 32, 32, 18, 4, 0); // fwd3
        launch_pack(convs[wb + 6], convs[wb + 7], apack[8 * s + 3], 32, 32, 18, 4, 0); // fwd4
        launch_pack(convs[wb + 6], convs[wb + 7], apack[8 * s + 4], 32, 32, 18, 4, 1); // T4
        launch_pack(convs[wb + 4], convs[wb + 5], apack[8 * s + 5], 32, 32, 18, 4, 1); // T3
        launch_pack(convs[wb + 2], convs[wb + 3], apack[8 * s + 6], 32, 32, 18, 4, 1); // T2
        launch_pack(convs[wb + 0], convs[wb + 1], apack[8 * s + 7], 32,  1, 18, 1, 1); // T1
    }

    const dim3 cgrid(NP / 16 / 8);   // 1800 blocks x 8 waves = 14400 N-tiles
    auto conv_1_32 = [&](const float* ir, const float* ii, const __bf16* ap,
                         float* orr, float* oi, int act) {
        conv_wmma_kernel<1, 1, 4><<<cgrid, 256, 4 * 1 * 512 * 2, stream>>>(
            ir, ii, ap, orr, oi, 32, act);
    };
    auto conv_32_32 = [&](const float* ir, const float* ii, const __bf16* ap,
                          float* orr, float* oi, int act) {
        conv_wmma_kernel<32, 18, 4><<<cgrid, 256, 4 * 18 * 512 * 2, stream>>>(
            ir, ii, ap, orr, oi, 32, act);
    };
    auto conv_32_1 = [&](const float* ir, const float* ii, const __bf16* ap,
                         float* orr, float* oi, int act) {
        conv_wmma_kernel<32, 18, 1><<<cgrid, 256, 1 * 18 * 512 * 2, stream>>>(
            ir, ii, ap, orr, oi, 1, act);
    };
    auto launch_cgemm = [&](const float* Ar, const float* Ai,
                            const float* Br_, const float* Bi_,
                            float* Cr, float* Ci,
                            int M, int K, int N, int as_, int bs_, int cs_) {
        dim3 grid((N + 15) / 16, (M + 15) / 16, BB);
        cgemm_kernel<<<grid, dim3(16, 16), 0, stream>>>(Ar, Ai, Br_, Bi_, Cr, Ci,
                                                        M, K, N, as_, bs_, cs_);
    };

    // ---- 3 phases x 2 streams ----
    for (int ph = 0; ph < 3; ++ph) {
        for (int s = 0; s < 2; ++s) {
            float* xr = s ? x2r : x1r;
            float* xi = s ? x2i : x1i;
            const float* kre = s ? k2r : k1r;
            const float* kim = s ? k2i : k1i;
            __bf16** ap = apack + 8 * s;
            const float* thr = s ? thr2 : thr1;
            const float* al  = s ? al2 : al1;
            const float* be  = s ? be2 : be1;

            // grad_f -> (t2r,t2i):  IFH * ((FH * x * FW) o m - k) o m * IFW
            launch_cgemm(FHr, FHi, xr, xi, t1r, t1i, HH, HH, WW, 0, HW, HW);
            launch_cgemm(t1r, t1i, FWr, FWi, t2r, t2i, HH, WW, WW, HW, 0, HW);
            maskres_kernel<<<NP / 256, 256, 0, stream>>>(t2r, t2i, mask, kre, kim, NP);
            launch_cgemm(IFHr, IFHi, t2r, t2i, t1r, t1i, HH, HH, WW, 0, HW, HW);
            launch_cgemm(t1r, t1i, IFWr, IFWi, t2r, t2i, HH, WW, WW, HW, 0, HW);

            // grad_r -> (grr,gri)
            conv_1_32 (xr,  xi,  ap[0], a1r, a1i, 1);
            conv_32_32(a1r, a1i, ap[1], a2r, a2i, 1);
            conv_32_32(a2r, a2i, ap[2], a3r, a3i, 1);
            conv_32_32(a3r, a3i, ap[3], w4r, w4i, 0);
            norm_scale_kernel<<<NP / 256, 256, 0, stream>>>(w4r, w4i, g0r, g0i, thr, NP);
            conv_32_32(g0r, g0i, ap[4], g1r, g1i, 0);
            actder_mul_kernel<<<CAP / 256, 256, 0, stream>>>(g1r, g1i, a3r, a3i, CAP);
            conv_32_32(g1r, g1i, ap[5], g0r, g0i, 0);
            actder_mul_kernel<<<CAP / 256, 256, 0, stream>>>(g0r, g0i, a2r, a2i, CAP);
            conv_32_32(g0r, g0i, ap[6], g1r, g1i, 0);
            actder_mul_kernel<<<CAP / 256, 256, 0, stream>>>(g1r, g1i, a1r, a1i, CAP);
            conv_32_1 (g1r, g1i, ap[7], grr, gri, 0);

            // x <- x - alpha*grad_f - beta*grad_r   (both grads from old x)
            update_kernel<<<NP / 256, 256, 0, stream>>>(xr, xi, t2r, t2i, grr, gri,
                                                        al, be, ph, NP);
        }
    }

    pack_out_kernel<<<NP / 256, 256, 0, stream>>>(x1r, x1i, x2r, x2i, (float*)d_out, NP);
}
